// STCrossAttnBlock_44547400794816
// MI455X (gfx1250) — matrix operand order repaired and checked by reading, not compiled, not run on hardware
//
#include <hip/hip_runtime.h>

// ---------------------------------------------------------------------------
// Problem constants (from reference): x:(12,512,48,48) f32
//   b=4 batches, 3 frames, c=512, n=48*48=2304, m=3n=6912, Mtot=4*6912=27648
// ---------------------------------------------------------------------------
#define C_DIM 512
#define N1    2304
#define N3    6912
#define MT    27648
#define SCORE_SCALE 0.04419417382415922f   // 512^-0.5

typedef __bf16 bf16_t;
typedef __attribute__((ext_vector_type(16))) bf16_t v16bf;
typedef __attribute__((ext_vector_type(8)))  float  v8f;

__device__ __forceinline__ v8f wmma_bf16(v16bf a, v16bf b, v8f c) {
  // D = A(16x32) * B(32x16) + C, fp32 accumulate
  return __builtin_amdgcn_wmma_f32_16x16x32_bf16(false, a, false, b, (short)0, c,
                                                 false, false);
}

// A fragment: 16(M) x 32(K) tile from row-major memory, leading dim ld (elems).
// Per ISA 16-bit A layout: lanes 0-15 -> M=lane, K = {0..7, 16..23};
//                          lanes 16-31 -> M=lane-16, K = {8..15, 24..31}.
__device__ __forceinline__ v16bf load_a(const bf16_t* base, int ld) {
  int lane = threadIdx.x & 31;
  int m    = lane & 15;
  int hi   = lane >> 4;
  const bf16_t* r = base + (long)m * ld + hi * 8;
  union { v16bf v; uint4 q[2]; } u;
  u.q[0] = *reinterpret_cast<const uint4*>(r);
  u.q[1] = *reinterpret_cast<const uint4*>(r + 16);
  return u.v;
}

// B fragment: 32(K) x 16(N) tile, loaded from B^T stored row-major (N rows, K cols).
// Same K order as A: lanes 0-15 -> N=lane, K = {0..7, 16..23}; lanes 16-31 -> K = {8..15, 24..31}.
__device__ __forceinline__ v16bf load_b(const bf16_t* baseT, int ld) {
  int lane = threadIdx.x & 31;
  int n    = lane & 15;
  int hi   = lane >> 4;
  const bf16_t* r = baseT + (long)n * ld + hi * 8;
  union { v16bf v; uint4 q[2]; } u;
  u.q[0] = *reinterpret_cast<const uint4*>(r);
  u.q[1] = *reinterpret_cast<const uint4*>(r + 16);
  return u.v;
}

// ---------------------------------------------------------------------------
// GroupNorm stats: 16 blocks, one per (b,g). Reduces 128*3*2304 elements.
// stats[grp*2] = mean, stats[grp*2+1] = rstd
// ---------------------------------------------------------------------------
__global__ __launch_bounds__(256) void gn_stats(const float* __restrict__ x,
                                                float* __restrict__ stats) {
  int grp = blockIdx.x;            // grp = b*4 + g
  int b = grp >> 2, g = grp & 3;
  const long total = 128L * 3L * N1;   // 884736
  float sum = 0.f, sq = 0.f;
  for (long e = threadIdx.x; e < total; e += 256) {
    int  s  = (int)(e % N1);
    long t  = e / N1;
    int  f  = (int)(t % 3);
    int  cp = (int)(t / 3);
    float v = x[((long)(f * 4 + b) * C_DIM + (g * 128 + cp)) * N1 + s];
    sum += v; sq += v * v;
  }
  __shared__ float rs[256], rq[256];
  int tid = threadIdx.x;
  rs[tid] = sum; rq[tid] = sq;
  __syncthreads();
  for (int st = 128; st > 0; st >>= 1) {
    if (tid < st) { rs[tid] += rs[tid + st]; rq[tid] += rq[tid + st]; }
    __syncthreads();
  }
  if (tid == 0) {
    float inv  = 1.f / (float)total;
    float mean = rs[0] * inv;
    float var  = rq[0] * inv - mean * mean;
    stats[grp * 2]     = mean;
    stats[grp * 2 + 1] = rsqrtf(var + 1e-6f);
  }
}

// ---------------------------------------------------------------------------
// GroupNorm apply + transpose -> hnT[(b*6912 + f*2304 + s), c] bf16
// Block handles a 32(s) x 32(c) tile via LDS transpose.
// grid = (16 cblk, 864 rblk)
// ---------------------------------------------------------------------------
__global__ __launch_bounds__(256) void gn_norm(const float* __restrict__ x,
                                               const float* __restrict__ stats,
                                               const float* __restrict__ nscale,
                                               const float* __restrict__ nbias,
                                               bf16_t* __restrict__ hnT) {
  __shared__ float tile[32][33];
  int c0   = blockIdx.x * 32;
  int row0 = blockIdx.y * 32;              // row = b*6912 + f*2304 + s
  int b    = row0 / N3;
  int rem  = row0 - b * N3;
  int f    = rem / N1;
  int s0   = rem - f * N1;
  int tid  = threadIdx.x;
  int ls   = tid & 31;                     // spatial within tile (coalesced read)
  int lc8  = tid >> 5;                     // 0..7
#pragma unroll
  for (int p = 0; p < 4; ++p) {
    int c = c0 + lc8 + p * 8;
    float v = x[((long)(f * 4 + b) * C_DIM + c) * N1 + s0 + ls];
    int g = c >> 7;
    float mean = stats[(b * 4 + g) * 2];
    float rstd = stats[(b * 4 + g) * 2 + 1];
    tile[ls][lc8 + p * 8] = (v - mean) * rstd * nscale[c] + nbias[c];
  }
  __syncthreads();
  int lc  = tid & 31;                      // channel within tile (coalesced write)
  int lr8 = tid >> 5;
#pragma unroll
  for (int p = 0; p < 4; ++p) {
    int r = lr8 + p * 8;
    hnT[(long)(row0 + r) * C_DIM + c0 + lc] = (bf16_t)tile[r][lc];
  }
}

// ---------------------------------------------------------------------------
// fp32 -> bf16 weight conversion
// ---------------------------------------------------------------------------
__global__ __launch_bounds__(256) void wcvt(const float* __restrict__ w,
                                            bf16_t* __restrict__ o, int n) {
  int i = blockIdx.x * 256 + threadIdx.x;
  if (i < n) o[i] = (bf16_t)w[i];
}

// ---------------------------------------------------------------------------
// Generic 1x1-conv GEMM:  D[r, n] = sum_c A[r, c] * W[n, c] + bias[n]
// Block = 8 waves; each wave computes a 16x64 output tile (4 WMMA N-tiles),
// K = 512 in 16 steps of 32. grid.x = Mrows/16.
// mode 0: kT[r*512 + n] = bf16(D)
// mode 1: vCM[n*MT + r] = bf16(D)                         (channel-major v)
// mode 2: (q, frame-1 rows) qT[r*512+n] = bf16(D); skip -> outF[b,n,s] = D
// mode 3: (proj) outF[b,n,s] += D                          (skip already there)
// ---------------------------------------------------------------------------
__global__ __launch_bounds__(256) void gemm16(const bf16_t* __restrict__ Ain,
                                              const bf16_t* __restrict__ Wb,
                                              const float* __restrict__ bias,
                                              bf16_t* __restrict__ outB,
                                              float* __restrict__ outF,
                                              int mode) {
  int r0   = blockIdx.x * 16;
  int wv   = threadIdx.x >> 5;
  int lane = threadIdx.x & 31;
  int n0   = wv * 64;

  const bf16_t* abase;
  if (mode == 2) {                         // q: only frame 1 rows of hnT
    int b = r0 / N1, s0 = r0 % N1;
    abase = Ain + (long)(b * N3 + N1 + s0) * C_DIM;
  } else {
    abase = Ain + (long)r0 * C_DIM;
  }

  const v8f zz = {0.f, 0.f, 0.f, 0.f, 0.f, 0.f, 0.f, 0.f};
  v8f acc[4] = {zz, zz, zz, zz};

  for (int k = 0; k < C_DIM; k += 32) {
    v16bf a = load_a(abase + k, C_DIM);
#pragma unroll
    for (int j = 0; j < 4; ++j) {
      v16bf bb = load_b(Wb + (long)(n0 + j * 16) * C_DIM + k, C_DIM);
      acc[j] = wmma_bf16(a, bb, acc[j]);
    }
  }

  int hi = lane >> 4, nl = lane & 15;
#pragma unroll
  for (int j = 0; j < 4; ++j) {
    int n = n0 + j * 16 + nl;
    float bv = bias[n];
#pragma unroll
    for (int v = 0; v < 8; ++v) {
      int r = r0 + v + 8 * hi;
      float val = acc[j][v] + bv;
      if (mode == 0) {
        outB[(long)r * C_DIM + n] = (bf16_t)val;
      } else if (mode == 1) {
        outB[(long)n * MT + r] = (bf16_t)val;
      } else if (mode == 2) {
        int b = r / N1, s = r - b * N1;
        outB[(long)r * C_DIM + n] = (bf16_t)val;
        outF[((long)b * C_DIM + n) * N1 + s] = val;   // skip
      } else {
        int b = r / N1, s = r - b * N1;
        outF[((long)b * C_DIM + n) * N1 + s] += val;  // skip + proj
      }
    }
  }
}

// ---------------------------------------------------------------------------
// Flash attention. Block = one (b, 16-query tile), 8 waves.
// Scores: each wave computes partials over its 64-channel K-slice; partials
// reduced in LDS. Online softmax by 16 threads. Output: each wave owns a
// 64-wide slice of the 512-wide fp32 accumulator; P tile (bf16) lives in LDS.
// grid = (144, 4)
// ---------------------------------------------------------------------------
__global__ __launch_bounds__(256) void flash_attn(const bf16_t* __restrict__ qT,
                                                  const bf16_t* __restrict__ kT,
                                                  const bf16_t* __restrict__ vCM,
                                                  bf16_t* __restrict__ attnT) {
  int q0   = blockIdx.x * 16;
  int b    = blockIdx.y;
  int tid  = threadIdx.x;
  int wv   = tid >> 5;
  int lane = tid & 31;
  int hi   = lane >> 4, nl = lane & 15;
  int c0   = wv * 64;

  __shared__ float Spart[8][16][32];
  __shared__ float Ssum[16][32];
  __shared__ __align__(16) bf16_t Prob[16][32];
  __shared__ float fac[16];
  __shared__ float lsh[16];

  float m_i = -1e30f, l_i = 0.f;     // meaningful in threads tid < 16

  const v8f zz = {0.f, 0.f, 0.f, 0.f, 0.f, 0.f, 0.f, 0.f};
  v8f oacc[4] = {zz, zz, zz, zz};

  const bf16_t* qbase = qT + (long)(b * N1 + q0) * C_DIM;

  for (int m0 = 0; m0 < N3; m0 += 32) {
    // ---- partial scores over this wave's 64-channel slice ----
    v8f sacc[2] = {zz, zz};
#pragma unroll
    for (int ks = 0; ks < 2; ++ks) {
      int k = c0 + ks * 32;
      v16bf a = load_a(qbase + k, C_DIM);
#pragma unroll
      for (int nt = 0; nt < 2; ++nt) {
        v16bf bb = load_b(kT + (long)(b * N3 + m0 + nt * 16) * C_DIM + k, C_DIM);
        sacc[nt] = wmma_bf16(a, bb, sacc[nt]);
      }
    }
#pragma unroll
    for (int nt = 0; nt < 2; ++nt)
#pragma unroll
      for (int v = 0; v < 8; ++v)
        Spart[wv][v + 8 * hi][nt * 16 + nl] = sacc[nt][v];
    __syncthreads();

    // ---- cross-wave reduction of partial scores (+ 1/sqrt(c) scale) ----
    for (int e = tid; e < 512; e += 256) {
      int r = e >> 5, cc = e & 31;
      float s = 0.f;
#pragma unroll
      for (int w = 0; w < 8; ++w) s += Spart[w][r][cc];
      Ssum[r][cc] = s * SCORE_SCALE;
    }
    __syncthreads();

    // ---- online softmax (one thread per query row) ----
    if (tid < 16) {
      float sv[32];
      float cmax = -1e30f;
#pragma unroll
      for (int j = 0; j < 32; ++j) { sv[j] = Ssum[tid][j]; cmax = fmaxf(cmax, sv[j]); }
      float mnew = fmaxf(m_i, cmax);
      float f    = __expf(m_i - mnew);
      float ssum = 0.f;
#pragma unroll
      for (int j = 0; j < 32; ++j) {
        float p = __expf(sv[j] - mnew);
        ssum += p;
        Prob[tid][j] = (bf16_t)p;
      }
      l_i = l_i * f + ssum;
      m_i = mnew;
      fac[tid] = f;
    }
    __syncthreads();

    // ---- rescale running accumulators, then oacc += P @ V^T slice ----
    float fv[8];
#pragma unroll
    for (int v = 0; v < 8; ++v) fv[v] = fac[v + 8 * hi];
#pragma unroll
    for (int j = 0; j < 4; ++j)
#pragma unroll
      for (int v = 0; v < 8; ++v) oacc[j][v] *= fv[v];

    v16bf p = load_a(&Prob[0][0], 32);
    const bf16_t* vb = vCM + (long)c0 * MT + (b * N3 + m0);
#pragma unroll
    for (int j = 0; j < 4; ++j) {
      v16bf bb = load_b(vb + (long)(j * 16) * MT, MT);
      oacc[j] = wmma_bf16(p, bb, oacc[j]);
    }
    __syncthreads();   // protect Prob/Spart before next chunk
  }

  if (tid < 16) lsh[tid] = l_i;
  __syncthreads();
  float linv[8];
#pragma unroll
  for (int v = 0; v < 8; ++v) linv[v] = 1.f / lsh[v + 8 * hi];
#pragma unroll
  for (int j = 0; j < 4; ++j)
#pragma unroll
    for (int v = 0; v < 8; ++v)
      attnT[(long)(b * N1 + q0 + v + 8 * hi) * C_DIM + c0 + j * 16 + nl] =
          (bf16_t)(oacc[j][v] * linv[v]);
}

// ---------------------------------------------------------------------------
// Host-side launch
// inputs: x, norm_scale, norm_bias, q_w, q_b, k_w, k_b, v_w, v_b, proj_w, proj_b
// output: (4, 512, 48, 48) fp32
// ---------------------------------------------------------------------------
extern "C" void kernel_launch(void* const* d_in, const int* in_sizes, int n_in,
                              void* d_out, int out_size, void* d_ws, size_t ws_size,
                              hipStream_t stream) {
  (void)in_sizes; (void)n_in; (void)out_size; (void)ws_size;
  const float* x   = (const float*)d_in[0];
  const float* nsc = (const float*)d_in[1];
  const float* nbi = (const float*)d_in[2];
  const float* qw  = (const float*)d_in[3];
  const float* qb  = (const float*)d_in[4];
  const float* kw  = (const float*)d_in[5];
  const float* kb  = (const float*)d_in[6];
  const float* vw  = (const float*)d_in[7];
  const float* vb  = (const float*)d_in[8];
  const float* pw  = (const float*)d_in[9];
  const float* pb  = (const float*)d_in[10];
  float* out = (float*)d_out;

  char* ws = (char*)d_ws;
  size_t off = 0;
  auto carve = [&](size_t bytes) -> void* {
    off = (off + 255) & ~(size_t)255;
    void* p = ws + off;
    off += bytes;
    return p;
  };
  float*  stats = (float*)carve(16 * 2 * sizeof(float));
  bf16_t* hnT   = (bf16_t*)carve((size_t)MT * C_DIM * sizeof(bf16_t));
  bf16_t* kTm   = (bf16_t*)carve((size_t)MT * C_DIM * sizeof(bf16_t));
  bf16_t* vCM   = (bf16_t*)carve((size_t)C_DIM * MT * sizeof(bf16_t));
  bf16_t* qTm   = (bf16_t*)carve((size_t)4 * N1 * C_DIM * sizeof(bf16_t));
  bf16_t* attnT = (bf16_t*)carve((size_t)4 * N1 * C_DIM * sizeof(bf16_t));
  bf16_t* wqb   = (bf16_t*)carve((size_t)C_DIM * C_DIM * sizeof(bf16_t));
  bf16_t* wkb   = (bf16_t*)carve((size_t)C_DIM * C_DIM * sizeof(bf16_t));
  bf16_t* wvb   = (bf16_t*)carve((size_t)C_DIM * C_DIM * sizeof(bf16_t));
  bf16_t* wpb   = (bf16_t*)carve((size_t)C_DIM * C_DIM * sizeof(bf16_t));

  gn_stats<<<16, 256, 0, stream>>>(x, stats);
  gn_norm<<<dim3(16, 864), 256, 0, stream>>>(x, stats, nsc, nbi, hnT);

  const int WN = C_DIM * C_DIM;           // 262144
  wcvt<<<WN / 256, 256, 0, stream>>>(qw, wqb, WN);
  wcvt<<<WN / 256, 256, 0, stream>>>(kw, wkb, WN);
  wcvt<<<WN / 256, 256, 0, stream>>>(vw, wvb, WN);
  wcvt<<<WN / 256, 256, 0, stream>>>(pw, wpb, WN);

  gemm16<<<MT / 16, 256, 0, stream>>>(hnT, wkb, kb, kTm, nullptr, 0);   // k
  gemm16<<<MT / 16, 256, 0, stream>>>(hnT, wvb, vb, vCM, nullptr, 1);   // v (c-major)
  gemm16<<<(4 * N1) / 16, 256, 0, stream>>>(hnT, wqb, qb, qTm, out, 2); // q + skip

  flash_attn<<<dim3(N1 / 16, 4), 256, 0, stream>>>(qTm, kTm, vCM, attnT);

  gemm16<<<(4 * N1) / 16, 256, 0, stream>>>(attnT, wpb, pb, nullptr, out, 3); // proj += skip
}
